// MultiHeadAttention_20246475833674
// MI455X (gfx1250) — compile-verified, matrix operand order checked by reading, not run on hardware
//
#include <hip/hip_runtime.h>

#define NB 4
#define LQ 2048
#define LK 2048
#define HH 16
#define PP 64
#define DD 1024
#define NTOK (NB * LQ)  // 8192
#define RCHUNK 64       // BN partial-reduction row chunks (128 rows each)

typedef __attribute__((ext_vector_type(16))) _Float16 v16h;
typedef __attribute__((ext_vector_type(8)))  float    v8f;

// ---------------------------------------------------------------------------
// Fragment loaders. CDNA5 WMMA 16x16x32 f16 layout (CDNA5 ISA 7.12.2):
//   A (16x32, MxK): lane L holds row M = L&15; element e maps to
//       k = (L>>4)*8 + (e>>3)*16 + (e&7)
//   B (32x16, KxN): mirror of A with N = L&15 and same k pattern.
//   C/D (16x16 f32): VGPR r holds row M = r + 8*(L>=16), col N = L&15.
// Caller passes a pointer to this lane's row/col at k0 + (L>>4)*8 so every
// even/odd element pair is a contiguous, 4-byte-aligned dword (the compiler
// further merges these into global_load_b128 clauses).
// ---------------------------------------------------------------------------
__device__ __forceinline__ v16h frag_from_f32(const float* __restrict__ p) {
  v16h a;
#pragma unroll
  for (int e = 0; e < 16; e += 2) {
    int k = ((e >> 3) << 4) + (e & 7);
    float2 f = *(const float2*)(p + k);
    a[e] = (_Float16)f.x;
    a[e + 1] = (_Float16)f.y;
  }
  return a;
}

__device__ __forceinline__ v16h frag_from_f16(const _Float16* __restrict__ p) {
  v16h a;
#pragma unroll
  for (int e = 0; e < 16; e += 2) {
    int k = ((e >> 3) << 4) + (e & 7);
    union { unsigned u; _Float16 h[2]; } c;
    c.u = *(const unsigned*)(p + k);
    a[e] = c.h[0];
    a[e + 1] = c.h[1];
  }
  return a;
}

// ---------------------------------------------------------------------------
// Kernel 1: fused QKV projection. Y = X @ W^T, one 16x64 tile per wave:
// the A fragment is reused across 4 B-column tiles (4 accumulators), so each
// 32-k chunk costs 1 A-frag + 4 B-frag loads for 4 WMMAs.
// mat 0: Q = query@Wq^T (f16 row-major)
// mat 1: K = key  @Wk^T (f16 row-major)
// mat 2: V = key  @Wv^T stored TRANSPOSED as Vt[((b*H+h)*P+p)*LK + l]
//        (n0 is 64-aligned => the 64-col block is exactly one head)
// ---------------------------------------------------------------------------
__global__ void qkv_proj_kernel(const float* __restrict__ query,
                                const float* __restrict__ key,
                                const float* __restrict__ Wq,
                                const float* __restrict__ Wk,
                                const float* __restrict__ Wv,
                                _Float16* __restrict__ Qh,
                                _Float16* __restrict__ Kh,
                                _Float16* __restrict__ Vt) {
  const int lane = threadIdx.x & 31;
  const int wid = (blockIdx.x * blockDim.x + threadIdx.x) >> 5;
  const int TPM = (NTOK / 16) * (DD / 64);  // 8192 tiles per matrix

  int mat = wid / TPM;
  int t = wid - mat * TPM;
  int tm = t / (DD / 64);
  int tn = t - tm * (DD / 64);
  const int m0 = tm * 16, n0 = tn * 64;

  const float* X = (mat == 0) ? query : key;
  const float* W = (mat == 0) ? Wq : (mat == 1 ? Wk : Wv);

  const int nn = lane & 15;
  const int hi8 = (lane >> 4) << 3;

  const float* arow = X + (size_t)(m0 + nn) * DD + hi8;
  const float* brow = W + (size_t)(n0 + nn) * DD + hi8;  // B[k][n] = W[n][k]

  v8f acc[4] = {};
#pragma unroll 2
  for (int k0 = 0; k0 < DD; k0 += 32) {
    v16h a = frag_from_f32(arow + k0);
#pragma unroll
    for (int j = 0; j < 4; ++j) {
      v16h b = frag_from_f32(brow + (size_t)j * 16 * DD + k0);
      acc[j] = __builtin_amdgcn_wmma_f32_16x16x32_f16(false, a, false, b,
                                                      (short)0, acc[j],
                                                      false, false);
    }
  }

  if (mat == 2) {
    const int h = n0 >> 6;  // whole 64-col block belongs to one head
#pragma unroll
    for (int j = 0; j < 4; ++j) {
      const int p = j * 16 + nn;
#pragma unroll
      for (int r = 0; r < 8; ++r) {
        int g = m0 + r + hi8;  // global token row
        int b = g >> 11, l = g & (LK - 1);
        Vt[(((size_t)b * HH + h) * PP + p) * LK + l] = (_Float16)acc[j][r];
      }
    }
  } else {
    _Float16* Y = (mat == 0) ? Qh : Kh;
#pragma unroll
    for (int j = 0; j < 4; ++j)
#pragma unroll
      for (int r = 0; r < 8; ++r) {
        int g = m0 + r + hi8;
        Y[(size_t)g * DD + n0 + j * 16 + nn] = (_Float16)acc[j][r];
      }
  }
}

// ---------------------------------------------------------------------------
// Kernel 2: causal flash attention. One wave per (batch, head, 16-query tile).
// Q tile (16x64 f16) stays in registers; keys processed 32 at a time:
//   4 score WMMAs -> scale -> causal mask -> online softmax
//   P re-laid out C->A through per-wave LDS (ds path)
//   4 AV WMMAs into f32 accumulators (P = 64 = 4 x 16-col C tiles)
// Next key block's K/V lines are prefetched (global_prefetch_b8) so the
// softmax VALU/trans work overlaps the HBM fetch.
// ---------------------------------------------------------------------------
__global__ void attn_kernel(const _Float16* __restrict__ Qh,
                            const _Float16* __restrict__ Kh,
                            const _Float16* __restrict__ Vt,
                            const float* __restrict__ query,
                            float* __restrict__ res) {
  __shared__ _Float16 lds[4][16 * 32];

  const int lane = threadIdx.x & 31;
  const int w = threadIdx.x >> 5;
  const int wid = (blockIdx.x * blockDim.x + threadIdx.x) >> 5;

  const int qt = wid & (LQ / 16 - 1);  // 128 query tiles
  const int nh = wid >> 7;
  const int h = nh & (HH - 1);
  const int nb = nh >> 4;
  const int q0 = qt * 16;

  const int nn = lane & 15;
  const int hi8 = (lane >> 4) << 3;

  // Q fragments for k-chunks {0..31, 32..63} of P=64
  const _Float16* qrow = Qh + (size_t)(nb * LQ + q0 + nn) * DD + h * PP + hi8;
  const v16h Qa0 = frag_from_f16(qrow);
  const v16h Qa1 = frag_from_f16(qrow + 32);

  const _Float16* kbase = Kh + (size_t)(nb * LK + nn) * DD + h * PP + hi8;
  const _Float16* vbase = Vt + (((size_t)nb * HH + h) * PP + nn) * LK + hi8;

  float m[8], l[8];
#pragma unroll
  for (int r = 0; r < 8; ++r) { m[r] = -1e30f; l[r] = 0.0f; }
  v8f O[4] = {};

  const int nkb = (q0 >> 5) + 1;  // causally-live 32-key blocks
  for (int kb = 0; kb < nkb; ++kb) {
    // prefetch next key block (K rows + V columns) while we compute this one
    if (kb + 1 < nkb) {
      __builtin_prefetch(kbase + (size_t)((kb + 1) * 32) * DD, 0, 0);
      __builtin_prefetch(kbase + (size_t)((kb + 1) * 32 + 16) * DD, 0, 0);
      __builtin_prefetch(vbase + (kb + 1) * 32, 0, 0);
      __builtin_prefetch(vbase + (size_t)32 * LK + (kb + 1) * 32, 0, 0);
    }

    v8f S[2];
#pragma unroll
    for (int sub = 0; sub < 2; ++sub) {
      const int kt = kb * 32 + sub * 16;
      const _Float16* krow = kbase + (size_t)kt * DD;
      v16h B0 = frag_from_f16(krow);
      v16h B1 = frag_from_f16(krow + 32);
      v8f c = {};
      c = __builtin_amdgcn_wmma_f32_16x16x32_f16(false, Qa0, false, B0,
                                                 (short)0, c, false, false);
      c = __builtin_amdgcn_wmma_f32_16x16x32_f16(false, Qa1, false, B1,
                                                 (short)0, c, false, false);
      const int kg = kt + nn;
#pragma unroll
      for (int r = 0; r < 8; ++r) {
        int qg = q0 + r + hi8;
        float s = c[r] * 0.03125f;  // 1/sqrt(D) = 1/32
        S[sub][r] = (kg > qg) ? -1000000.0f : s;
      }
    }

    // online softmax: row reductions across the 16-lane group holding a row
    float alpha[8];
#pragma unroll
    for (int r = 0; r < 8; ++r) {
      float v = fmaxf(S[0][r], S[1][r]);
#pragma unroll
      for (int off = 1; off < 16; off <<= 1)
        v = fmaxf(v, __shfl_xor(v, off, 32));
      float mn = fmaxf(m[r], v);
      alpha[r] = __expf(m[r] - mn);
      m[r] = mn;
    }
#pragma unroll
    for (int r = 0; r < 8; ++r) {
      float p0 = __expf(S[0][r] - m[r]);
      float p1 = __expf(S[1][r] - m[r]);
      S[0][r] = p0;
      S[1][r] = p1;
      float v = p0 + p1;
#pragma unroll
      for (int off = 1; off < 16; off <<= 1)
        v += __shfl_xor(v, off, 32);
      l[r] = l[r] * alpha[r] + v;
    }
#pragma unroll
    for (int pc = 0; pc < 4; ++pc)
#pragma unroll
      for (int r = 0; r < 8; ++r)
        O[pc][r] *= alpha[r];

    // C-layout -> A-layout transpose of P through per-wave LDS (16x32 f16)
    _Float16* pl = lds[w];
#pragma unroll
    for (int sub = 0; sub < 2; ++sub)
#pragma unroll
      for (int r = 0; r < 8; ++r)
        pl[(r + hi8) * 32 + sub * 16 + nn] = (_Float16)S[sub][r];
    v16h Pa = frag_from_f16(pl + (lane & 15) * 32 + hi8);

    // AV: B[kk][p] from transposed V => contiguous along kk
#pragma unroll
    for (int pc = 0; pc < 4; ++pc) {
      v16h Vb = frag_from_f16(vbase + (size_t)(pc * 16) * LK + kb * 32);
      O[pc] = __builtin_amdgcn_wmma_f32_16x16x32_f16(false, Pa, false, Vb,
                                                     (short)0, O[pc],
                                                     false, false);
    }
  }

  // epilogue: normalize by row sum, add residual, store f32
  const float* qres = query + (size_t)(nb * LQ + q0) * DD + h * PP;
  float* orow = res + (size_t)(nb * LQ + q0) * DD + h * PP;
#pragma unroll
  for (int pc = 0; pc < 4; ++pc) {
#pragma unroll
    for (int r = 0; r < 8; ++r) {
      int g = r + hi8;
      int c = pc * 16 + nn;
      orow[(size_t)g * DD + c] = O[pc][r] / l[r] + qres[(size_t)g * DD + c];
    }
  }
}

// ---------------------------------------------------------------------------
// Kernel 3a: BN partial sums. 64 row-chunks x 1024 columns, coalesced.
// 64 blocks x 256 threads (512 waves) so the 32MB sweep can use HBM bandwidth.
// ---------------------------------------------------------------------------
__global__ void bn_partial_kernel(const float* __restrict__ res,
                                  float* __restrict__ psum,
                                  float* __restrict__ psumsq) {
  const int chunk = blockIdx.x;
  const int r0 = chunk * (NTOK / RCHUNK);
#pragma unroll
  for (int c = 0; c < DD / 256; ++c) {
    const int d = threadIdx.x + c * 256;
    float s = 0.0f, ss = 0.0f;
    for (int r = 0; r < NTOK / RCHUNK; ++r) {
      float x = res[(size_t)(r0 + r) * DD + d];
      s += x;
      ss += x * x;
    }
    psum[chunk * DD + d] = s;
    psumsq[chunk * DD + d] = ss;
  }
}

// ---------------------------------------------------------------------------
// Kernel 3b: finalize mean / rstd per column.
// ---------------------------------------------------------------------------
__global__ void bn_finalize_kernel(const float* __restrict__ psum,
                                   const float* __restrict__ psumsq,
                                   float* __restrict__ mean,
                                   float* __restrict__ rstd) {
  const int d = blockIdx.x * blockDim.x + threadIdx.x;
  float s = 0.0f, ss = 0.0f;
#pragma unroll 8
  for (int c = 0; c < RCHUNK; ++c) {
    s += psum[c * DD + d];
    ss += psumsq[c * DD + d];
  }
  const float inv = 1.0f / (float)NTOK;
  float mu = s * inv;
  float var = ss * inv - mu * mu;
  mean[d] = mu;
  rstd[d] = rsqrtf(var + 1e-5f);
}

// ---------------------------------------------------------------------------
// Kernel 4: BN normalize + affine, float4-vectorized (b128 traffic).
// ---------------------------------------------------------------------------
__global__ void bn_apply_kernel(const float* __restrict__ res,
                                const float* __restrict__ mean,
                                const float* __restrict__ rstd,
                                const float* __restrict__ gamma,
                                const float* __restrict__ beta,
                                float* __restrict__ out) {
  size_t i4 = ((size_t)blockIdx.x * blockDim.x + threadIdx.x) * 4;
  int d = (int)(i4 & (DD - 1));
  float4 x = *(const float4*)(res + i4);
  float4 mu = *(const float4*)(mean + d);
  float4 rs = *(const float4*)(rstd + d);
  float4 ga = *(const float4*)(gamma + d);
  float4 be = *(const float4*)(beta + d);
  float4 y;
  y.x = (x.x - mu.x) * rs.x * ga.x + be.x;
  y.y = (x.y - mu.y) * rs.y * ga.y + be.y;
  y.z = (x.z - mu.z) * rs.z * ga.z + be.z;
  y.w = (x.w - mu.w) * rs.w * ga.w + be.w;
  *(float4*)(out + i4) = y;
}

// ---------------------------------------------------------------------------
extern "C" void kernel_launch(void* const* d_in, const int* in_sizes, int n_in,
                              void* d_out, int out_size, void* d_ws, size_t ws_size,
                              hipStream_t stream) {
  (void)in_sizes; (void)n_in; (void)out_size; (void)ws_size;

  const float* query = (const float*)d_in[0];
  const float* key   = (const float*)d_in[1];
  const float* Wq    = (const float*)d_in[2];
  const float* Wk    = (const float*)d_in[3];
  const float* Wv    = (const float*)d_in[4];
  const float* gamma = (const float*)d_in[5];
  const float* beta  = (const float*)d_in[6];
  float* out = (float*)d_out;

  char* ws = (char*)d_ws;
  _Float16* Qh = (_Float16*)ws; ws += (size_t)NTOK * DD * 2;      // 16 MB
  _Float16* Kh = (_Float16*)ws; ws += (size_t)NTOK * DD * 2;      // 16 MB
  _Float16* Vt = (_Float16*)ws; ws += (size_t)NTOK * DD * 2;      // 16 MB
  float* res    = (float*)ws;   ws += (size_t)NTOK * DD * 4;      // 32 MB
  float* psum   = (float*)ws;   ws += (size_t)RCHUNK * DD * 4;    // 256 KB
  float* psumsq = (float*)ws;   ws += (size_t)RCHUNK * DD * 4;    // 256 KB
  float* mean   = (float*)ws;   ws += (size_t)DD * 4;
  float* rstd   = (float*)ws;

  // 1) QKV projections: 3 * 512 * 16 tiles (16x64 each), 4 waves/block
  const int tiles = 3 * (NTOK / 16) * (DD / 64);  // 24576 waves
  qkv_proj_kernel<<<tiles / 4, 128, 0, stream>>>(query, key, Wq, Wk, Wv,
                                                 Qh, Kh, Vt);
  // 2) attention: NB*H*(LQ/16) = 8192 waves
  const int awaves = NB * HH * (LQ / 16);
  attn_kernel<<<awaves / 4, 128, 0, stream>>>(Qh, Kh, Vt, query, res);
  // 3) BN stats: partial + finalize
  bn_partial_kernel<<<RCHUNK, 256, 0, stream>>>(res, psum, psumsq);
  bn_finalize_kernel<<<DD / 256, 256, 0, stream>>>(psum, psumsq, mean, rstd);
  // 4) BN apply (float4 per thread)
  bn_apply_kernel<<<(int)(((size_t)NTOK * DD) / (4 * 256)), 256, 0, stream>>>(
      res, mean, rstd, gamma, beta, out);
}
